// CSA_66408784331435
// MI455X (gfx1250) — matrix operand order repaired and checked by reading, not compiled, not run on hardware
//
#include <hip/hip_runtime.h>
#include <hip/hip_bf16.h>
#include <cstdint>
#include <cstddef>

typedef __attribute__((ext_vector_type(16))) __bf16 v16bf;
typedef __attribute__((ext_vector_type(8)))  float  v8f;

#define LL 4096
#define CC 64
#define BB 4

union FragU { uint4 u[2]; v16bf v; };

// fp32 -> bf16 bits, round-to-nearest-even
__device__ __forceinline__ unsigned short f2bf(float x) {
    unsigned u = __builtin_bit_cast(unsigned, x);
    unsigned r = u + 0x7FFFu + ((u >> 16) & 1u);
    return (unsigned short)(r >> 16);
}

// A fragment (16x32 bf16, M x K): row-major [row][K] source.
// Lane (g = lane>>4): halves i=0..15 at K = kbase + g*16 + i.
__device__ __forceinline__ v16bf frag_a(const unsigned short* base, int rowstride,
                                        int row, int kbase, int g) {
    FragU f;
    const unsigned short* p = base + row * rowstride + kbase + g * 16;
    f.u[0] = *(const uint4*)(p);
    f.u[1] = *(const uint4*)(p + 8);
    return f.v;
}

// B fragment (32x16 bf16, K x N) from a K-contiguous (transposed) store [n][K].
// Lane: n = given, halves i=0..15 at K = kbase + g*16 + i (one 32B contiguous read).
__device__ __forceinline__ v16bf frag_b(const unsigned short* base, int rowstride,
                                        int n, int kbase, int g) {
    FragU f;
    const uint4* p = (const uint4*)(base + n * rowstride + kbase + g * 16);
    f.u[0] = p[0];
    f.u[1] = p[1];
    return f.v;
}

// Async 16-byte copy global -> LDS via CDNA5 async-LDS path (tracked by ASYNCcnt).
// LDS byte address = low 32 bits of the flat shared pointer (group-segment offset).
__device__ __forceinline__ void async_cp16(const unsigned short* gsrc,
                                           unsigned short* ldst) {
    unsigned lds = (unsigned)(unsigned long long)(ldst);
    unsigned long long ga = (unsigned long long)(gsrc);
    asm volatile("global_load_async_to_lds_b128 %0, %1, off"
                 :: "v"(lds), "v"(ga) : "memory");
}
__device__ __forceinline__ void wait_async0() {
    asm volatile("s_wait_asynccnt 0" ::: "memory");
}

// ---------------------------------------------------------------------------
// Pass 1: conv1d (K=3, pad=1) as GEMM  M=64(co) x K=192(t,ci) x N=128(l tile)
// Output stored transposed bf16: yT[b][l][co]
// ---------------------------------------------------------------------------
__global__ __launch_bounds__(256) void conv1d_wmma(
    const float* __restrict__ x,      // [B][64][L]
    const float* __restrict__ w,      // [64][64][3]
    const float* __restrict__ bias,   // [64]
    unsigned short* __restrict__ yT)  // [B][L][64] bf16 bits
{
    __shared__ __align__(16) unsigned short xT[130 * 64];   // rows l0-1 .. l0+128, [row][ci]
    __shared__ __align__(16) unsigned short wA[64 * 192];   // [co][t*64+ci]
    __shared__ float bsh[64];

    const int tid = threadIdx.x;
    const int b   = blockIdx.y;
    const int l0  = blockIdx.x * 128;

    for (int i = tid; i < 130 * 64; i += 256) {
        int ci = i / 130;
        int r  = i % 130;
        int l  = l0 - 1 + r;
        float v = (l >= 0 && l < LL) ? x[((size_t)b * CC + ci) * LL + l] : 0.0f;
        xT[r * 64 + ci] = f2bf(v);
    }
    for (int i = tid; i < 64 * 192; i += 256) {
        int co = i / 192, k = i % 192;
        int t = k / 64, ci = k % 64;
        wA[i] = f2bf(w[(co * 64 + ci) * 3 + t]);
    }
    if (tid < 64) bsh[tid] = bias[tid];
    __syncthreads();

    const int lane  = tid & 31;
    const int wv    = tid >> 5;
    const int g     = lane >> 4;
    const int co16  = wv & 3;
    const int jbase = (wv >> 2) * 4;

    v8f acc[4];
    for (int j = 0; j < 4; ++j)
        for (int r = 0; r < 8; ++r) acc[j][r] = 0.0f;

    for (int ks = 0; ks < 6; ++ks) {   // K = 192 = 6 x 32
        v16bf a = frag_a(wA, 192, co16 * 16 + (lane & 15), ks * 32, g);
        for (int j = 0; j < 4; ++j) {
            int lrow = (jbase + j) * 16 + (lane & 15);
            int kk = ks * 32 + g * 16;           // 16-aligned, within one tap
            int t  = kk >> 6, ci = kk & 63;
            FragU f;
            const uint4* p = (const uint4*)(xT + (lrow + t) * 64 + ci);
            f.u[0] = p[0];
            f.u[1] = p[1];
            acc[j] = __builtin_amdgcn_wmma_f32_16x16x32_bf16(
                false, a, false, f.v, (short)0, acc[j], false, false);
        }
    }

    for (int j = 0; j < 4; ++j) {
        int l   = l0 + (jbase + j) * 16 + (lane & 15);
        int cob = co16 * 16 + g * 8;
        unsigned short t8[8];
        for (int r = 0; r < 8; ++r) t8[r] = f2bf(acc[j][r] + bsh[cob + r]);
        uint4 pk;
        pk.x = t8[0] | ((unsigned)t8[1] << 16);
        pk.y = t8[2] | ((unsigned)t8[3] << 16);
        pk.z = t8[4] | ((unsigned)t8[5] << 16);
        pk.w = t8[6] | ((unsigned)t8[7] << 16);
        *(uint4*)(yT + ((size_t)b * LL + l) * 64 + cob) = pk;
    }
}

// ---------------------------------------------------------------------------
// Pass 2: fused  S = relu(K^T Q)/sqrt(3) ;  out = V * S
// Double-buffered l-tiles (32), async global->LDS staging, 2 barriers/iter.
// ---------------------------------------------------------------------------
__global__ __launch_bounds__(256) void attn_wmma(
    const unsigned short* __restrict__ kT,  // [B][L][64] bf16
    const unsigned short* __restrict__ qT,  // [B][L][64] bf16
    const unsigned short* __restrict__ vT,  // [B][L][64] bf16
    float* __restrict__ out)                // [B][64][L] fp32
{
    __shared__ __align__(16) unsigned short qS[128 * 64];     // [m][c]
    __shared__ __align__(16) unsigned short kS[2][32 * 64];   // [l][c] double-buffered
    __shared__ __align__(16) unsigned short vS[2][64 * 32];   // [c][l] double-buffered
    __shared__ __align__(16) unsigned short sS[128 * 32];     // [m][l]

    const int tid  = threadIdx.x;
    const int b    = blockIdx.y;
    const int m0   = blockIdx.x * 128;
    const int lane = tid & 31;
    const int wv   = tid >> 5;
    const int g    = lane >> 4;
    const size_t bbase = (size_t)b * LL * CC;   // element offset of batch

    // qS preload via async-LDS (1024 x 16B chunks, 4 per thread)
    for (int i = tid; i < 128 * 64 / 8; i += 256)
        async_cp16(qT + bbase + (size_t)m0 * 64 + i * 8, qS + i * 8);

    const int lrel  = tid >> 3;         // vS stage: row 0..31
    const int cbase = (tid & 7) * 8;    //           col 0..56

    // Prologue: stage tile 0 into buffer 0
    async_cp16(kT + bbase + tid * 8, kS[0] + tid * 8);   // 256 x 16B = whole kS tile
    {
        uint4 d = *(const uint4*)(vT + bbase + (size_t)lrel * 64 + cbase);
        const unsigned short* ds = (const unsigned short*)&d;
        for (int e = 0; e < 8; ++e) vS[0][(cbase + e) * 32 + lrel] = ds[e];
    }
    wait_async0();
    __syncthreads();

    const int li = wv & 1;    // GEMM1: l 16-block
    const int mp = wv >> 1;   // GEMM1: m block pair
    const int cb = wv & 3;    // GEMM2: c 16-block
    const int mh = wv >> 2;   // GEMM2: m 64-half

    v8f oacc[4];
    for (int j = 0; j < 4; ++j)
        for (int r = 0; r < 8; ++r) oacc[j][r] = 0.0f;

    const float scale = 0.5773502691896258f;  // 1/sqrt(3)
    const int NIT = LL / 32;

    for (int lt = 0; lt < NIT; ++lt) {
        const int cur = lt & 1, nxt = cur ^ 1;
        const bool pf = (lt + 1) < NIT;     // uniform
        uint4 vreg;

        // Prefetch tile lt+1: kS via async-LDS, vS into registers
        if (pf) {
            const int l1 = (lt + 1) * 32;
            async_cp16(kT + bbase + (size_t)l1 * 64 + tid * 8, kS[nxt] + tid * 8);
            vreg = *(const uint4*)(vT + bbase + (size_t)(l1 + lrel) * 64 + cbase);
        }

        // GEMM1: s[l,m] = sum_c kT[l][c] * qT[m][c]
        for (int jj = 0; jj < 2; ++jj) {
            int mj = mp * 2 + jj;
            v8f s;
            for (int r = 0; r < 8; ++r) s[r] = 0.0f;
            for (int ks = 0; ks < 2; ++ks) {
                v16bf a  = frag_a(kS[cur], 64, li * 16 + (lane & 15), ks * 32, g);
                v16bf bb = frag_b(qS, 64, mj * 16 + (lane & 15), ks * 32, g);
                s = __builtin_amdgcn_wmma_f32_16x16x32_bf16(
                    false, a, false, bb, (short)0, s, false, false);
            }
            int m  = mj * 16 + (lane & 15);
            int lb = li * 16 + g * 8;
            unsigned short t8[8];
            for (int r = 0; r < 8; ++r) {
                float v = s[r];
                v = (v > 0.0f) ? v * scale : 0.0f;
                t8[r] = f2bf(v);
            }
            uint4 pk;
            pk.x = t8[0] | ((unsigned)t8[1] << 16);
            pk.y = t8[2] | ((unsigned)t8[3] << 16);
            pk.z = t8[4] | ((unsigned)t8[5] << 16);
            pk.w = t8[6] | ((unsigned)t8[7] << 16);
            *(uint4*)(sS + m * 32 + lb) = pk;
        }
        __syncthreads();   // sS visible; all kS[cur] reads done

        // GEMM2: out[c,m] += sum_l v[c,l] * s[l,m]
        {
            v16bf a = frag_a(vS[cur], 32, cb * 16 + (lane & 15), 0, g);
            for (int j = 0; j < 4; ++j) {
                v16bf bb = frag_b(sS, 32, mh * 64 + j * 16 + (lane & 15), 0, g);
                oacc[j] = __builtin_amdgcn_wmma_f32_16x16x32_bf16(
                    false, a, false, bb, (short)0, oacc[j], false, false);
            }
        }

        // Complete the prefetch into the next buffers
        if (pf) {
            const unsigned short* ds = (const unsigned short*)&vreg;
            for (int e = 0; e < 8; ++e) vS[nxt][(cbase + e) * 32 + lrel] = ds[e];
            wait_async0();
        }
        __syncthreads();   // next buffers visible; sS reads done
    }

    // Epilogue: write out[b][c][m] fp32
    for (int j = 0; j < 4; ++j) {
        int m  = m0 + mh * 64 + j * 16 + (lane & 15);
        int c0 = cb * 16 + g * 8;
        for (int r = 0; r < 8; ++r) {
            out[((size_t)b * CC + c0 + r) * LL + m] = oacc[j][r];
        }
    }
}

extern "C" void kernel_launch(void* const* d_in, const int* in_sizes, int n_in,
                              void* d_out, int out_size, void* d_ws, size_t ws_size,
                              hipStream_t stream) {
    (void)in_sizes; (void)n_in; (void)out_size; (void)ws_size;

    const float* x  = (const float*)d_in[0];
    const float* Kw = (const float*)d_in[1];
    const float* Kb = (const float*)d_in[2];
    const float* Qw = (const float*)d_in[3];
    const float* Qb = (const float*)d_in[4];
    const float* Vw = (const float*)d_in[5];
    const float* Vb = (const float*)d_in[6];
    float* out = (float*)d_out;

    // Workspace: kT | qT | vT, each [B][L][64] bf16 = 2 MiB (6 MiB total)
    unsigned short* kT = (unsigned short*)d_ws;
    unsigned short* qT = kT + (size_t)BB * LL * CC;
    unsigned short* vT = qT + (size_t)BB * LL * CC;

    dim3 grid1(LL / 128, BB);
    conv1d_wmma<<<grid1, 256, 0, stream>>>(x, Kw, Kb, kT);
    conv1d_wmma<<<grid1, 256, 0, stream>>>(x, Qw, Qb, qT);
    conv1d_wmma<<<grid1, 256, 0, stream>>>(x, Vw, Vb, vT);

    dim3 grid2(LL / 128, BB);
    attn_wmma<<<grid2, 256, 0, stream>>>(kT, qT, vT, out);
}
